// SMC_51539608008
// MI455X (gfx1250) — matrix operand order repaired and checked by reading, not compiled, not run on hardware
//
#include <hip/hip_runtime.h>

// SMC particle filter for MI455X (gfx1250).
// Kernel 1: full-GPU transpose z[K][T] -> zt[T][K] (workspace, 128MB).
// Kernel 2: single persistent workgroup (1024 thr = 32 wave32) scans T steps.
//   - w[] double-buffered in LDS, cdf[] in LDS (~228KB of the 320KB/WGP)
//   - per-wave async global->LDS prefetch of next z,u rows (ASYNCcnt)
//   - per-wave 256-elem prefix scan via ONE v_wmma_f32_16x16x32_f16:
//       A = [L|L] (triangular ones, exact f16), B = [Xhi;Xlo] f16 split
//       => D = L*Xhi + L*Xlo accumulated in f32 (~f32 accuracy)
//   - all cross-lane reductions/scans as DPP16/permlanex16 VALU (no ds_bpermute)

#define K_P   8192
#define T_S   4096
#define BLK   1024
#define PPT   8          // particles per thread
#define NWAVE 32

typedef __attribute__((ext_vector_type(16))) _Float16 v16h;
typedef __attribute__((ext_vector_type(8)))  float    v8f;

// ---- DPP / permlane helpers (all-VALU cross-lane ops) ----
#define ROW_SHR(n)   (0x110 + (n))
#define ROW_SHARE(n) (0x150 + (n))
#define ROW_XMASK(n) (0x160 + (n))

template <int CTRL>
__device__ __forceinline__ float dpp_add_src(float v) {
  // shifted copy with additive identity (0) for invalid lanes
  return __int_as_float(__builtin_amdgcn_update_dpp(
      0, __float_as_int(v), CTRL, 0xf, 0xf, true));
}
template <int CTRL>
__device__ __forceinline__ float dpp_mov(float v) {
  int i = __float_as_int(v);
  return __int_as_float(__builtin_amdgcn_update_dpp(i, i, CTRL, 0xf, 0xf, false));
}
__device__ __forceinline__ float permx16_id(float v) {  // lane i <-> other-row lane i
  int i = __float_as_int(v);
  return __int_as_float(__builtin_amdgcn_permlanex16(i, i, 0x76543210u, 0xfedcba98u, false, false));
}
__device__ __forceinline__ float permx16_b15(float v) { // all lanes <- other-row lane 15
  int i = __float_as_int(v);
  return __int_as_float(__builtin_amdgcn_permlanex16(i, i, 0xffffffffu, 0xffffffffu, false, false));
}
__device__ __forceinline__ float wave_max32(float m) {
  m = fmaxf(m, dpp_mov<ROW_XMASK(1)>(m));
  m = fmaxf(m, dpp_mov<ROW_XMASK(2)>(m));
  m = fmaxf(m, dpp_mov<ROW_XMASK(4)>(m));
  m = fmaxf(m, dpp_mov<ROW_XMASK(8)>(m));
  m = fmaxf(m, permx16_id(m));
  return m;                                // all 32 lanes hold the max
}
__device__ __forceinline__ float row_incl_scan16(float v) {
  v += dpp_add_src<ROW_SHR(1)>(v);
  v += dpp_add_src<ROW_SHR(2)>(v);
  v += dpp_add_src<ROW_SHR(4)>(v);
  v += dpp_add_src<ROW_SHR(8)>(v);
  return v;                                // inclusive scan within each 16-lane row
}

__device__ __forceinline__ unsigned lds_addr32(const void* p) {
  // shared-memory flat address: low 32 bits are the LDS byte offset
  return (unsigned)(unsigned long long)p;
}

__device__ __forceinline__ void async_wait0() {
#if __has_builtin(__builtin_amdgcn_s_wait_asynccnt)
  __builtin_amdgcn_s_wait_asynccnt(0);
#else
  asm volatile("s_wait_asynccnt 0" ::: "memory");
#endif
}

// copy 1KB (256 floats): 32 lanes x b128 x 2 instructions, wave-private.
__device__ __forceinline__ void async_copy_1k(const float* gsrc, float* ldsdst) {
  unsigned l = lds_addr32(ldsdst);
  asm volatile(
      "global_load_async_to_lds_b128 %0, %1, off\n\t"
      "global_load_async_to_lds_b128 %0, %1, off offset:512"
      :: "v"(l), "v"(gsrc) : "memory");
}

__global__ void __launch_bounds__(256)
transpose_z(const float* __restrict__ z, float* __restrict__ zt) {
  __shared__ float tile[32][33];
  const int t0 = blockIdx.x << 5, k0 = blockIdx.y << 5;
  const int tx = threadIdx.x, ty = threadIdx.y;
#pragma unroll
  for (int j = 0; j < 4; ++j)
    tile[ty + j * 8][tx] = z[(size_t)(k0 + ty + j * 8) * T_S + t0 + tx];
  __syncthreads();
#pragma unroll
  for (int j = 0; j < 4; ++j)
    zt[(size_t)(t0 + ty + j * 8) * K_P + k0 + tx] = tile[tx][ty + j * 8];
}

__global__ void __launch_bounds__(BLK)
smc_scan(const float* __restrict__ x, const float* __restrict__ w0,
         const float* __restrict__ u, const float* __restrict__ zt,
         float* __restrict__ out) {
  __shared__ float zbuf[2][K_P];     // async-prefetched z rows (wave-private slices)
  __shared__ float ubuf[2][K_P];     // async-prefetched u rows (wave-private slices)
  __shared__ float wbuf[2][K_P];     // particle states, double-buffered
  __shared__ float cdf[K_P];         // p staging, then scanned CDF
  __shared__ float wavesums[NWAVE];
  __shared__ float waveoff[NWAVE];
  __shared__ float red[NWAVE];
  __shared__ float sM, sS;

  const int tid   = threadIdx.x;
  const int lane  = tid & 31;
  const int wid   = tid >> 5;
  const int m16   = lane & 15;       // WMMA A-row (M) == B-col (N) for this lane
  const int g     = lane >> 4;       // half-wave group
  const int wbase = wid << 8;        // wave's 256-particle tile base
  const int kb    = tid * PPT;       // this thread's particles (natural mapping)
  const int soff  = wbase + (lane << 2);  // this lane's async-slice offset

  // init particle states
#pragma unroll
  for (int j = 0; j < PPT; ++j) wbuf[0][kb + j] = w0[kb + j];

  // constant A = [L | L], L[i][j] = (j <= i), exact in f16.
  // A 16x32 f16 layout: lane m16, group g: v16h[j] = A[m16][K], K = 16g + j, Kmod16 = j
  v16h amat;
#pragma unroll
  for (int j = 0; j < 16; ++j)
    amat[j] = (_Float16)((j <= m16) ? 1.0f : 0.0f);

  const float LOGK     = 9.010913347279289f;    // log(8192)
  const float NHLOG2PI = -0.9189385332046727f;  // -0.5*log(2*pi)
  float lme_acc = 0.0f;

  // prefetch row 0 of z and u (wave-private 1KB slices)
  async_copy_1k(zt + soff, &zbuf[0][soff]);
  async_copy_1k(u  + soff, &ubuf[0][soff]);

  for (int t = 0; t < T_S; ++t) {
    const int cb = t & 1;
    const float xt = x[t];                    // uniform -> scalar load
    async_wait0();                            // my wave's z,u rows t resident

    float zv[PPT], wv[PPT], lw[PPT];
#pragma unroll
    for (int j = 0; j < PPT; ++j) zv[j] = zbuf[cb][kb + j];

    if (t + 1 < T_S) {                        // overlap next-row fetch with this step
      const size_t nrow = (size_t)(t + 1) * K_P + soff;
      async_copy_1k(zt + nrow, &zbuf[cb ^ 1][soff]);
      async_copy_1k(u  + nrow, &ubuf[cb ^ 1][soff]);
    }

#pragma unroll
    for (int j = 0; j < PPT; ++j) wv[j] = wbuf[cb][kb + j];

    // log_ws = -0.5*(z-w)^2 - 0.5*log2pi + x*z - 0.5*x^2
    const float c0 = fmaf(-0.5f * xt, xt, NHLOG2PI);
    float lmax = -3.4e38f;
#pragma unroll
    for (int j = 0; j < PPT; ++j) {
      const float d = zv[j] - wv[j];
      const float v = fmaf(-0.5f, d * d, fmaf(xt, zv[j], c0));
      lw[j] = v;
      lmax = fmaxf(lmax, v);
    }
    // block max (logsumexp stabilization): DPP wave max, LDS for cross-wave
    lmax = wave_max32(lmax);
    if (lane == 0) red[wid] = lmax;
    __syncthreads();
    if (tid < 32) {
      float v = wave_max32(red[lane]);
      if (lane == 0) sM = v;
    }
    __syncthreads();
    const float mval = sM;

    // p = exp(lw - m); stage into cdf[] (same-wave region, in-order LDS)
#pragma unroll
    for (int j = 0; j < PPT; ++j) lw[j] = __expf(lw[j] - mval);
    {
      float4 p0 = {lw[0], lw[1], lw[2], lw[3]};
      float4 p1 = {lw[4], lw[5], lw[6], lw[7]};
      ((float4*)&cdf[kb])[0] = p0;
      ((float4*)&cdf[kb])[1] = p1;
    }

    // ---- wave-level 256-element inclusive scan via WMMA ----
    // B 32x16 f16 layout: lane (n=m16, g): v16h[j] = B[16g + j][n]
    //   rows 0..15 = Xhi, rows 16..31 = Xlo, X[i][n] = p[wbase + n*16 + i]
    float pv[16];
    {
      const float4* p4 = (const float4*)&cdf[wbase + (m16 << 4)];
      float4 q;
      q = p4[0]; pv[0]  = q.x; pv[1]  = q.y; pv[2]  = q.z; pv[3]  = q.w;
      q = p4[1]; pv[4]  = q.x; pv[5]  = q.y; pv[6]  = q.z; pv[7]  = q.w;
      q = p4[2]; pv[8]  = q.x; pv[9]  = q.y; pv[10] = q.z; pv[11] = q.w;
      q = p4[3]; pv[12] = q.x; pv[13] = q.y; pv[14] = q.z; pv[15] = q.w;
    }
    v16h bmat;
#pragma unroll
    for (int j = 0; j < 16; ++j) {
      const float hi = (float)(_Float16)pv[j];
      bmat[j] = (_Float16)(g ? (pv[j] - hi) : pv[j]);
    }
    v8f acc = {0.f, 0.f, 0.f, 0.f, 0.f, 0.f, 0.f, 0.f};
    acc = __builtin_amdgcn_wmma_f32_16x16x32_f16(false, amat, false, bmat,
                                                 (short)0, acc, false, false);
    // acc[r] = inclusive scan at tile position m16*16 + 8g + r (8 consecutive)

    // cross-segment offsets: segsum[n] = D[15][n] lives in lane 16+n, vreg 7
    const float xr = permx16_id(acc[7]);       // other-row acc[7], lane-matched
    const float ss = g ? acc[7] : xr;          // segsum[m16] in every lane
    const float incl = row_incl_scan16(ss);    // DPP scan (rows identical)
    const float segoff = incl - ss;
    const float wsum = dpp_mov<ROW_SHARE(15)>(incl);  // wave-tile total
    if (lane == 0) wavesums[wid] = wsum;
    __syncthreads();
    if (tid < 32) {                       // cross-wave exclusive scan of 32 sums
      const float v = wavesums[lane];
      float inc2 = row_incl_scan16(v);
      const float o15 = permx16_b15(inc2);
      if (lane >= 16) inc2 += o15;        // stitch the two rows
      waveoff[lane] = inc2 - v;
      if (lane == 31) sS = inc2;          // total unnormalized mass S
    }
    __syncthreads();
    const float S = sS;
    const float base = waveoff[wid] + segoff;
    {
      const int gpos = wbase + (m16 << 4) + (g << 3);
      float4 o0 = {acc[0] + base, acc[1] + base, acc[2] + base, acc[3] + base};
      float4 o1 = {acc[4] + base, acc[5] + base, acc[6] + base, acc[7] + base};
      ((float4*)&cdf[gpos])[0] = o0;
      ((float4*)&cdf[gpos])[1] = o1;
    }
    if (tid == 0) lme_acc += mval + __logf(S) - LOGK;
    __syncthreads();                      // full CDF visible

    // ---- resample: lower_bound(cdf, u*S) + gather ----
    const float4* ur = (const float4*)&ubuf[cb][kb];
    const float4 u0 = ur[0], u1 = ur[1];
    const float uv[8] = {u0.x, u0.y, u0.z, u0.w, u1.x, u1.y, u1.z, u1.w};
    float wn[PPT];
#pragma unroll
    for (int j = 0; j < PPT; ++j) {
      const float val = uv[j] * S;
      int pos = 0;
#pragma unroll
      for (int sh = 12; sh >= 0; --sh) {
        const int mid = pos + (1 << sh);
        if (cdf[mid - 1] < val) pos = mid; // pos <= 8191 by construction
      }
      wn[j] = wbuf[cb][pos];
    }
#pragma unroll
    for (int j = 0; j < PPT; ++j) wbuf[cb ^ 1][kb + j] = wn[j];
    __syncthreads();
  }

  if (tid == 0) out[0] = lme_acc;
}

extern "C" void kernel_launch(void* const* d_in, const int* in_sizes, int n_in,
                              void* d_out, int out_size, void* d_ws, size_t ws_size,
                              hipStream_t stream) {
  const float* x  = (const float*)d_in[0];   // [T]
  const float* w0 = (const float*)d_in[1];   // [K]
  const float* z  = (const float*)d_in[2];   // [K,T]
  const float* u  = (const float*)d_in[3];   // [T,K]
  float* out = (float*)d_out;
  float* zt  = (float*)d_ws;                 // [T,K] transposed z (128 MB)

  dim3 tgrid(T_S / 32, K_P / 32), tblk(32, 8);
  transpose_z<<<tgrid, tblk, 0, stream>>>(z, zt);
  smc_scan<<<1, BLK, 0, stream>>>(x, w0, u, zt, out);
}
